// GNN_68367289418047
// MI455X (gfx1250) — matrix operand order repaired and checked by reading, not compiled, not run on hardware
//
#include <hip/hip_runtime.h>

// ---------------------------------------------------------------------------
// CDNA5 (gfx1250) GNN forward: 5 layers of gather/segment-sum + MLP + BN.
// Bandwidth-bound (~1 GB/layer at 23.3 TB/s). GEMMs use wave32
// v_wmma_f32_16x16x32_bf16 (f32 accumulate). Weight panels are staged once
// per block into LDS (conflict-free padded stride) so the per-WMMA critical
// path sees only short-latency ds_load_b128; the A stream is register
// double-buffered and prefetched. Activations staged as bf16 so the
// N x 320 / N x 640 operands (64/128 MB) stay L2-resident (192 MB L2).
// ---------------------------------------------------------------------------

typedef __bf16 bf16;
typedef __attribute__((ext_vector_type(16))) __bf16 v16bf;
typedef __attribute__((ext_vector_type(8)))  __bf16 v8bf;
typedef __attribute__((ext_vector_type(8)))  float  v8f;

#define D_IN   300
#define KP1    320   // K of GEMM1, padded (300 -> 320)
#define NC1    640   // cols of GEMM1 output, padded (600 -> 640)
#define KP2    640   // K of GEMM2 (= NC1)
#define NC2    320   // cols of GEMM2 output, padded (300 -> 320)
#define NLAYER 5
#define LDSPAD 8     // +16B per row: 16 lanes -> 16 distinct 4-bank groups

static __device__ __forceinline__ unsigned short f32_to_bf16_bits(float f) {
  unsigned u = __float_as_uint(f);
  unsigned r = u + 0x7FFFu + ((u >> 16) & 1u);   // round-to-nearest-even
  return (unsigned short)(r >> 16);
}

static __device__ __forceinline__ v8f wmma_bf16(v16bf a, v16bf b, v8f c) {
  // 8 args: (neg_a, A, neg_b, B, c_mod, C, reuse_a, reuse_b)
  return __builtin_amdgcn_wmma_f32_16x16x32_bf16(false, a, false, b,
                                                 (short)0, c, false, false);
}

// ---------------------------------------------------------------------------
// Weight conversion (once, all layers): transpose + pad + f32->bf16.
// W1 (L,300,600) -> W1t (L,640,320): row j holds K for output column j.
// ---------------------------------------------------------------------------
__global__ void gnn_conv_w1(const float* __restrict__ W1,
                            unsigned short* __restrict__ W1t, int total) {
  int idx = blockIdx.x * blockDim.x + threadIdx.x;
  if (idx >= total) return;
  int l = idx / (NC1 * KP1);
  int rem = idx - l * (NC1 * KP1);
  int j = rem / KP1;            // output column 0..639
  int k = rem - j * KP1;        // K index 0..319
  float v = (k < 300 && j < 600) ? W1[(size_t)l * 300 * 600 + (size_t)k * 600 + j]
                                 : 0.0f;
  W1t[idx] = f32_to_bf16_bits(v);
}

// W2 (L,600,300) -> W2t (L,320,640)
__global__ void gnn_conv_w2(const float* __restrict__ W2,
                            unsigned short* __restrict__ W2t, int total) {
  int idx = blockIdx.x * blockDim.x + threadIdx.x;
  if (idx >= total) return;
  int l = idx / (NC2 * KP2);
  int rem = idx - l * (NC2 * KP2);
  int j = rem / KP2;            // output column 0..319
  int k = rem - j * KP2;        // K index 0..639
  float v = (k < 600 && j < 300) ? W2[(size_t)l * 600 * 300 + (size_t)k * 300 + j]
                                 : 0.0f;
  W2t[idx] = f32_to_bf16_bits(v);
}

// ---------------------------------------------------------------------------
// agg = h + self_emb   (self_emb = E1[l][4] + E2[l][0])
// ---------------------------------------------------------------------------
__global__ void gnn_prep(const float* __restrict__ h,
                         const float* __restrict__ E1l,
                         const float* __restrict__ E2l,
                         float* __restrict__ agg, int total) {
  int idx = blockIdx.x * blockDim.x + threadIdx.x;
  if (idx >= total) return;
  int d = idx % D_IN;
  agg[idx] = h[idx] + E1l[4 * D_IN + d] + E2l[d];
}

// ---------------------------------------------------------------------------
// Edge scatter: one wave32 per edge, f32 global atomics into agg[dst].
// ---------------------------------------------------------------------------
__global__ void gnn_scatter(const float* __restrict__ h,
                            const int* __restrict__ ei,   // (2,E)
                            const int* __restrict__ ea,   // (E,2)
                            const float* __restrict__ E1l,
                            const float* __restrict__ E2l,
                            float* __restrict__ agg, int E) {
  const int lane = threadIdx.x & 31;
  const int e = blockIdx.x * 8 + (threadIdx.x >> 5);
  if (e >= E) return;
  const int s  = ei[e];
  const int d  = ei[E + e];
  const int bt = ea[2 * e];
  const int bd = ea[2 * e + 1];
  const float* hs = h + (size_t)s * D_IN;
  const float* e1 = E1l + (size_t)bt * D_IN;
  const float* e2 = E2l + (size_t)bd * D_IN;
  float* ag = agg + (size_t)d * D_IN;
  for (int k = lane; k < D_IN; k += 32)
    atomicAdd(&ag[k], hs[k] + e1[k] + e2[k]);
}

// ---------------------------------------------------------------------------
// agg (N x 300 f32) -> aggb (N x 320 bf16, zero K-pad)
// ---------------------------------------------------------------------------
__global__ void gnn_cvt_agg(const float* __restrict__ agg,
                            unsigned short* __restrict__ aggb, int total) {
  int idx = blockIdx.x * blockDim.x + threadIdx.x;
  if (idx >= total) return;
  int k = idx % KP1;
  int n = idx / KP1;
  aggb[idx] = (k < 300) ? f32_to_bf16_bits(agg[(size_t)n * D_IN + k])
                        : (unsigned short)0;
}

// ---------------------------------------------------------------------------
// GEMM1: H = relu(Agg @ W1 + b1), bf16 in, bf16 out (N x 640).
// Block: 8 waves = 128 rows x 64 cols. B panel (64 x 320 bf16 = 40KB+pad)
// staged once into LDS; A register double-buffered.
// ---------------------------------------------------------------------------
__global__ __launch_bounds__(256) void gnn_gemm1(
    const bf16* __restrict__ A,        // N x 320
    const bf16* __restrict__ Bt,       // 640 x 320 (pre-transposed)
    const float* __restrict__ bias,    // 600 valid
    unsigned short* __restrict__ H,    // N x 640 (bf16 bits)
    int N) {
  constexpr int LDSW = KP1 + LDSPAD;
  __shared__ unsigned short sB[64 * LDSW];

  const int tid  = threadIdx.x;
  const int lane = tid & 31;
  const int wave = tid >> 5;
  const int row0 = blockIdx.y * 128 + wave * 16;
  const int col0 = blockIdx.x * 64;
  const int m  = lane & 15;
  const int kh = lane >> 4;

  // Cooperative B panel load: rows col0..col0+63 are contiguous in Bt.
  {
    const bf16* src = Bt + (size_t)col0 * KP1;
    for (int i = tid * 8; i < 64 * KP1; i += 256 * 8) {
      int j = i / KP1, k = i - j * KP1;            // KP1 % 8 == 0: no straddle
      *(v8bf*)&sB[j * LDSW + k] = *(const v8bf*)(src + i);
    }
  }
  __syncthreads();

  int ar = row0 + m; if (ar >= N) ar = N - 1;      // clamp loads, guard stores
  const bf16* ap = A + (size_t)ar * KP1 + kh * 16;
  const unsigned short* bp = &sB[m * LDSW + kh * 16];

  v8f c0 = {}, c1 = {}, c2 = {}, c3 = {};
  v8bf alo = *(const v8bf*)(ap);
  v8bf ahi = *(const v8bf*)(ap + 8);
#pragma unroll
  for (int kb = 0; kb < KP1; kb += 32) {
    const int kn = (kb + 32 < KP1) ? kb + 32 : 0;  // clamped next-step preload
    __builtin_prefetch(ap + kb + 96, 0, 0);
    v8bf nlo = *(const v8bf*)(ap + kn);
    v8bf nhi = *(const v8bf*)(ap + kn + 8);
    v16bf a = __builtin_shufflevector(alo, ahi, 0,1,2,3,4,5,6,7,8,9,10,11,12,13,14,15);
    v16bf b0 = *(const v16bf*)(bp + kb);
    v16bf b1 = *(const v16bf*)(bp + 16 * LDSW + kb);
    v16bf b2 = *(const v16bf*)(bp + 32 * LDSW + kb);
    v16bf b3 = *(const v16bf*)(bp + 48 * LDSW + kb);
    c0 = wmma_bf16(a, b0, c0);
    c1 = wmma_bf16(a, b1, c1);
    c2 = wmma_bf16(a, b2, c2);
    c3 = wmma_bf16(a, b3, c3);
    alo = nlo; ahi = nhi;
  }
#pragma unroll
  for (int t = 0; t < 4; ++t) {
    v8f c = (t == 0) ? c0 : (t == 1) ? c1 : (t == 2) ? c2 : c3;
    const int col = col0 + 16 * t + m;
    const float bj = (col < 600) ? bias[col] : 0.0f;
#pragma unroll
    for (int i = 0; i < 8; ++i) {
      const int r = row0 + i + 8 * kh;             // C layout: VGPR i -> row i + 8*kh
      if (r < N) {
        float v = c[i] + bj;
        v = v > 0.0f ? v : 0.0f;                   // fused ReLU
        H[(size_t)r * NC1 + col] = f32_to_bf16_bits(v);
      }
    }
  }
}

// ---------------------------------------------------------------------------
// GEMM2: M = H @ W2 + b2, bf16 in, f32 out (N x 320). B panel 64x640 (80KB+pad).
// ---------------------------------------------------------------------------
__global__ __launch_bounds__(256) void gnn_gemm2(
    const bf16* __restrict__ A,        // N x 640
    const bf16* __restrict__ Bt,       // 320 x 640 (pre-transposed)
    const float* __restrict__ bias,    // 300 valid
    float* __restrict__ M,             // N x 320
    int N) {
  constexpr int LDSW = KP2 + LDSPAD;
  __shared__ unsigned short sB[64 * LDSW];

  const int tid  = threadIdx.x;
  const int lane = tid & 31;
  const int wave = tid >> 5;
  const int row0 = blockIdx.y * 128 + wave * 16;
  const int col0 = blockIdx.x * 64;
  const int m  = lane & 15;
  const int kh = lane >> 4;

  {
    const bf16* src = Bt + (size_t)col0 * KP2;
    for (int i = tid * 8; i < 64 * KP2; i += 256 * 8) {
      int j = i / KP2, k = i - j * KP2;            // KP2 % 8 == 0: no straddle
      *(v8bf*)&sB[j * LDSW + k] = *(const v8bf*)(src + i);
    }
  }
  __syncthreads();

  int ar = row0 + m; if (ar >= N) ar = N - 1;
  const bf16* ap = A + (size_t)ar * KP2 + kh * 16;
  const unsigned short* bp = &sB[m * LDSW + kh * 16];

  v8f c0 = {}, c1 = {}, c2 = {}, c3 = {};
  v8bf alo = *(const v8bf*)(ap);
  v8bf ahi = *(const v8bf*)(ap + 8);
#pragma unroll 4
  for (int kb = 0; kb < KP2; kb += 32) {
    const int kn = (kb + 32 < KP2) ? kb + 32 : 0;
    __builtin_prefetch(ap + kb + 96, 0, 0);
    v8bf nlo = *(const v8bf*)(ap + kn);
    v8bf nhi = *(const v8bf*)(ap + kn + 8);
    v16bf a = __builtin_shufflevector(alo, ahi, 0,1,2,3,4,5,6,7,8,9,10,11,12,13,14,15);
    v16bf b0 = *(const v16bf*)(bp + kb);
    v16bf b1 = *(const v16bf*)(bp + 16 * LDSW + kb);
    v16bf b2 = *(const v16bf*)(bp + 32 * LDSW + kb);
    v16bf b3 = *(const v16bf*)(bp + 48 * LDSW + kb);
    c0 = wmma_bf16(a, b0, c0);
    c1 = wmma_bf16(a, b1, c1);
    c2 = wmma_bf16(a, b2, c2);
    c3 = wmma_bf16(a, b3, c3);
    alo = nlo; ahi = nhi;
  }
#pragma unroll
  for (int t = 0; t < 4; ++t) {
    v8f c = (t == 0) ? c0 : (t == 1) ? c1 : (t == 2) ? c2 : c3;
    const int col = col0 + 16 * t + m;
    const float bj = (col < 300) ? bias[col] : 0.0f;
#pragma unroll
    for (int i = 0; i < 8; ++i) {
      const int r = row0 + i + 8 * kh;
      if (r < N) M[(size_t)r * NC2 + col] = c[i] + bj;
    }
  }
}

// ---------------------------------------------------------------------------
// BatchNorm: zero stats / column-wise sum & sumsq / normalize + affine (+ReLU)
// ---------------------------------------------------------------------------
__global__ void gnn_zero(float* __restrict__ p, int n) {
  int idx = blockIdx.x * blockDim.x + threadIdx.x;
  if (idx < n) p[idx] = 0.0f;
}

__global__ __launch_bounds__(320) void gnn_bnstats(const float* __restrict__ M,
                                                   float* __restrict__ sums,
                                                   float* __restrict__ sumsq,
                                                   int N) {
  const int c = threadIdx.x;
  if (c >= 300) return;
  const int r0 = blockIdx.x * 256;
  const int r1 = (r0 + 256 < N) ? r0 + 256 : N;
  float s = 0.0f, s2 = 0.0f;
  for (int r = r0; r < r1; ++r) {
    float v = M[(size_t)r * NC2 + c];
    s += v;
    s2 += v * v;
  }
  atomicAdd(&sums[c], s);
  atomicAdd(&sumsq[c], s2);
}

__global__ void gnn_bnapply(const float* __restrict__ M,
                            const float* __restrict__ sums,
                            const float* __restrict__ sumsq,
                            const float* __restrict__ gamma,
                            const float* __restrict__ beta,
                            float* __restrict__ out,
                            int total, float invN, int do_relu) {
  int idx = blockIdx.x * blockDim.x + threadIdx.x;
  if (idx >= total) return;
  int d = idx % D_IN;
  int n = idx / D_IN;
  float mean = sums[d] * invN;
  float var  = sumsq[d] * invN - mean * mean;
  float v = (M[(size_t)n * NC2 + d] - mean) * rsqrtf(var + 1e-5f) * gamma[d] + beta[d];
  if (do_relu) v = v > 0.0f ? v : 0.0f;
  out[idx] = v;
}

// ---------------------------------------------------------------------------
// Host side
// ---------------------------------------------------------------------------
extern "C" void kernel_launch(void* const* d_in, const int* in_sizes, int n_in,
                              void* d_out, int out_size, void* d_ws, size_t ws_size,
                              hipStream_t stream) {
  const float* x     = (const float*)d_in[0];
  const int*   ei    = (const int*)d_in[1];
  const int*   ea    = (const int*)d_in[2];
  const float* W1    = (const float*)d_in[3];
  const float* b1    = (const float*)d_in[4];
  const float* W2    = (const float*)d_in[5];
  const float* b2    = (const float*)d_in[6];
  const float* E1    = (const float*)d_in[7];
  const float* E2    = (const float*)d_in[8];
  const float* gamma = (const float*)d_in[9];
  const float* beta  = (const float*)d_in[10];

  const int N = in_sizes[0] / D_IN;
  const int E = in_sizes[1] / 2;

  // Carve workspace (256B aligned regions).
  char* ws = (char*)d_ws;
  size_t off = 0;
  auto carve = [&](size_t bytes) -> void* {
    void* p = ws + off;
    off += (bytes + 255) & ~(size_t)255;
    return p;
  };
  float*          agg    = (float*)carve((size_t)N * D_IN * 4);
  unsigned short* aggb   = (unsigned short*)carve((size_t)N * KP1 * 2);
  unsigned short* hidden = (unsigned short*)carve((size_t)N * NC1 * 2);
  float*          mbuf   = (float*)carve((size_t)N * NC2 * 4);
  float*          hbuf   = (float*)carve((size_t)N * D_IN * 4);
  unsigned short* W1t    = (unsigned short*)carve((size_t)NLAYER * NC1 * KP1 * 2);
  unsigned short* W2t    = (unsigned short*)carve((size_t)NLAYER * NC2 * KP2 * 2);
  float*          sums   = (float*)carve(2 * NC2 * 4);
  float*          sumsq  = sums + NC2;

  // One-time weight transpose/pad/convert (tiny).
  {
    int t1 = NLAYER * NC1 * KP1;
    gnn_conv_w1<<<(t1 + 255) / 256, 256, 0, stream>>>(W1, W1t, t1);
    int t2 = NLAYER * NC2 * KP2;
    gnn_conv_w2<<<(t2 + 255) / 256, 256, 0, stream>>>(W2, W2t, t2);
  }

  const float* h = x;
  for (int l = 0; l < NLAYER; ++l) {
    const float* E1l = E1 + (size_t)l * 6 * D_IN;
    const float* E2l = E2 + (size_t)l * 3 * D_IN;

    // agg = h + self_emb
    int tprep = N * D_IN;
    gnn_prep<<<(tprep + 255) / 256, 256, 0, stream>>>(h, E1l, E2l, agg, tprep);

    // agg[dst] += h[src] + E1[bt] + E2[bd]
    gnn_scatter<<<(E + 7) / 8, 256, 0, stream>>>(h, ei, ea, E1l, E2l, agg, E);

    // f32 -> bf16 (K-padded)
    int tcvt = N * KP1;
    gnn_cvt_agg<<<(tcvt + 255) / 256, 256, 0, stream>>>(agg, aggb, tcvt);

    // hidden = relu(aggb @ W1 + b1)   [bf16 out]
    dim3 g1(NC1 / 64, (N + 127) / 128);
    gnn_gemm1<<<g1, 256, 0, stream>>>((const bf16*)aggb,
                                      (const bf16*)(W1t + (size_t)l * NC1 * KP1),
                                      b1 + (size_t)l * 600, hidden, N);

    // mbuf = hidden @ W2 + b2   [f32 out]
    dim3 g2(NC2 / 64, (N + 127) / 128);
    gnn_gemm2<<<g2, 256, 0, stream>>>((const bf16*)hidden,
                                      (const bf16*)(W2t + (size_t)l * NC2 * KP2),
                                      b2 + (size_t)l * 300, mbuf, N);

    // BatchNorm
    gnn_zero<<<(2 * NC2 + 255) / 256, 256, 0, stream>>>(sums, 2 * NC2);
    gnn_bnstats<<<(N + 255) / 256, 320, 0, stream>>>(mbuf, sums, sumsq, N);

    float* outp = (l == NLAYER - 1) ? (float*)d_out : hbuf;
    int tbn = N * D_IN;
    gnn_bnapply<<<(tbn + 255) / 256, 256, 0, stream>>>(
        mbuf, sums, sumsq, gamma + (size_t)l * D_IN, beta + (size_t)l * D_IN,
        outp, tbn, 1.0f / (float)N, (l < NLAYER - 1) ? 1 : 0);

    h = hbuf;
  }
}
